// DualResolutionAttention_74577812127847
// MI455X (gfx1250) — matrix operand order repaired and checked by reading, not compiled, not run on hardware
//
#include <hip/hip_runtime.h>
#include <hip/hip_bf16.h>

typedef __attribute__((ext_vector_type(16))) __bf16 v16bf;
typedef __attribute__((ext_vector_type(2)))  __bf16 v2bf;
typedef __attribute__((ext_vector_type(8)))  float  v8f;
typedef __attribute__((ext_vector_type(8)))  unsigned v8u;

#define DEV __device__ __forceinline__

static constexpr int B_  = 2;
static constexpr int T_  = 4096;
static constexpr int E_  = 1024;
static constexpr int HH_ = 8;     // heads per branch
static constexpr int HD_ = 64;    // head dim
static constexpr int LD_ = 512;
static constexpr int TC_ = 1024;  // T / R

// f32 -> bf16 (hardware RNE cast) as raw bits
DEV unsigned short f2bfu(float x) {
  return __builtin_bit_cast(unsigned short, (__bf16)x);
}
// pack two f32 into a dword of two bf16 -> single v_cvt_pk_bf16_f32
#if __has_builtin(__builtin_amdgcn_cvt_pk_bf16_f32)
DEV unsigned pack2(float a, float b) {
  return __builtin_bit_cast(unsigned, __builtin_amdgcn_cvt_pk_bf16_f32(a, b));
}
#else
DEV unsigned pack2(float a, float b) {
  return (unsigned)f2bfu(a) | ((unsigned)f2bfu(b) << 16);
}
#endif

// load a 16-half fragment from two contiguous 16-byte LDS runs
DEV v16bf ld_frag2(const unsigned short* p0, const unsigned short* p1) {
  uint4 a = *(const uint4*)p0;
  uint4 b = *(const uint4*)p1;
  v8u u;
  u[0] = a.x; u[1] = a.y; u[2] = a.z; u[3] = a.w;
  u[4] = b.x; u[5] = b.y; u[6] = b.z; u[7] = b.w;
  return __builtin_bit_cast(v16bf, u);
}

DEV v8f wmma_bf16(v16bf a, v16bf b, v8f c) {
  // D = A(16x32 bf16) * B(32x16 bf16) + C(16x16 f32)
  return __builtin_amdgcn_wmma_f32_16x16x32_bf16(
      /*neg_a=*/false, a, /*neg_b=*/false, b,
      /*c_mod=*/(short)0, c, /*reuse_a=*/false, /*reuse_b=*/false);
}

// ---------------------------------------------------------------------------
// GEMM:  C[M,N] = A[M,K] (row stride lda) * W[K,N] (row stride ldw) + bias[N]
// Block: 128 threads (4 waves). 64x64 tile; each wave owns 16 rows.
// M,N multiples of 64; K multiple of 64 (staged 64 per barrier interval).
// LDS: A tile [m][k]; W tile transposed [n][k] so B-fragments are contiguous.
// ---------------------------------------------------------------------------
__global__ void __launch_bounds__(128)
gemm_bias_kernel(const float* __restrict__ A, int lda,
                 const float* __restrict__ W, int ldw,
                 const float* __restrict__ bias,
                 float* __restrict__ C, int ldc,
                 int M, int N, int K)
{
  __shared__ __align__(16) unsigned short lA [64][72]; // [m][k],  144B rows
  __shared__ __align__(16) unsigned short lBt[64][72]; // [n][k],  144B rows

  const int tid  = threadIdx.x;
  const int wave = tid >> 5;
  const int lane = tid & 31;
  const int n15  = lane & 15;
  const int hg   = lane >> 4;      // lane half (wave32)
  const int m0   = blockIdx.y * 64;
  const int n0   = blockIdx.x * 64;

  v8f acc[4] = {};
  const int nkc = K >> 6;
  for (int kc = 0; kc < nkc; ++kc) {
    // stage A tile 64x64: 2048 packed dwords, coalesced float2 along k
    #pragma unroll
    for (int i = 0; i < 16; ++i) {
      int e  = tid + i * 128;          // 0..2047
      int am = e >> 5, ak = (e & 31) * 2;
      float2 av = *(const float2*)&A[(size_t)(m0 + am) * lda + kc * 64 + ak];
      *(unsigned*)&lA[am][ak] = pack2(av.x, av.y);
    }
    // stage W tile transposed 64x64: each thread reads one n from two
    // consecutive k rows (both lane-coalesced along n), packs one dword
    #pragma unroll
    for (int i = 0; i < 16; ++i) {
      int e  = tid + i * 128;          // 0..2047
      int kp = e >> 6, bn = e & 63;    // k-pair 0..31, col 0..63
      const float* wp = &W[(size_t)(kc * 64 + 2 * kp) * ldw + n0 + bn];
      *(unsigned*)&lBt[bn][2 * kp] = pack2(wp[0], wp[ldw]);
    }
    __syncthreads();

    #pragma unroll
    for (int k2 = 0; k2 < 2; ++k2) {
      // issue ALL fragment loads first so DS latency overlaps the WMMAs
      v16bf af = ld_frag2(&lA[wave * 16 + n15][k2 * 32 + hg * 8],
                          &lA[wave * 16 + n15][k2 * 32 + 16 + hg * 8]);
      v16bf bfr[4];
      #pragma unroll
      for (int nc = 0; nc < 4; ++nc) {
        const unsigned short* p = &lBt[nc * 16 + n15][k2 * 32 + hg * 8];
        bfr[nc] = ld_frag2(p, p + 16);
      }
      #pragma unroll
      for (int nc = 0; nc < 4; ++nc)
        acc[nc] = wmma_bf16(af, bfr[nc], acc[nc]);
    }
    __syncthreads();
  }

  // epilogue: C layout — VGPR r: lanes0-15 -> M=r, lanes16-31 -> M=r+8
  #pragma unroll
  for (int nc = 0; nc < 4; ++nc) {
    int n = n0 + nc * 16 + n15;
    float bv = bias[n];
    #pragma unroll
    for (int r = 0; r < 8; ++r) {
      int m = m0 + wave * 16 + hg * 8 + r;
      C[(size_t)m * ldc + n] = acc[nc][r] + bv;
    }
  }
}

// ---------------------------------------------------------------------------
// Causal flash attention over a fused qkv buffer (B*Tloc, 1536):
//   q = cols [0,512), k = [512,1024), v = [1024,1536); 8 heads x 64.
// Block: 128 threads = 4 waves sharing one 64-query tile; 32 keys per step.
// LDS: K row-major [key][d] (S-fragments contiguous over d),
//      V transposed [d][key] (O-fragments contiguous over key).
// ---------------------------------------------------------------------------
__global__ void __launch_bounds__(128)
flash_attn_kernel(const float* __restrict__ qkv, float* __restrict__ out, int Tloc)
{
  __shared__ __align__(16) unsigned short sK [32][72];    // [key][d], 144B rows
  __shared__ __align__(16) unsigned short sVt[64][40];    // [d][key], 80B rows
  __shared__ __align__(16) unsigned short sP [4][16][40]; // per-wave P [q][key]

  const int tid  = threadIdx.x;
  const int wave = tid >> 5;
  const int lane = tid & 31;
  const int n15  = lane & 15;
  const int hg   = lane >> 4;

  const int ntile = Tloc >> 6;
  const int tile  = blockIdx.x % ntile;
  const int h     = (blockIdx.x / ntile) % HH_;
  const int b     = blockIdx.x / (ntile * HH_);

  const int qcol = h * HD_;
  const int kcol = LD_ + h * HD_;
  const int vcol = 2 * LD_ + h * HD_;
  const size_t rowbase = (size_t)b * Tloc;

  // Q A-fragments (pre-scaled by 1/sqrt(64) = 0.125, exact in bf16)
  v16bf aq[2];
  {
    const size_t qrow = rowbase + tile * 64 + wave * 16 + n15;
    #pragma unroll
    for (int c = 0; c < 2; ++c) {
      v8u u;
      #pragma unroll
      for (int i = 0; i < 8; ++i) {
        int kk = c * 32 + ((i < 4) ? 0 : 16) + hg * 8 + (i & 3) * 2;
        float2 qv = *(const float2*)&qkv[qrow * 1536 + qcol + kk];
        u[i] = pack2(qv.x * 0.125f, qv.y * 0.125f);
      }
      aq[c] = __builtin_bit_cast(v16bf, u);
    }
  }

  float mstat[8], lstat[8];
  #pragma unroll
  for (int r = 0; r < 8; ++r) { mstat[r] = -1e30f; lstat[r] = 0.f; }
  v8f o[4] = {};

  const int nkb = tile * 2 + 2;  // keys 0 .. tile*64+63 (causal bound of tile)
  for (int kb = 0; kb < nkb; ++kb) {
    // stage K row-major: packed dword stores along d (coalesced float2)
    #pragma unroll
    for (int i = 0; i < 8; ++i) {
      int e   = tid + i * 128;          // 0..1023
      int key = e >> 5, d = (e & 31) * 2;
      size_t krow = rowbase + kb * 32 + key;
      float2 kv = *(const float2*)&qkv[krow * 1536 + kcol + d];
      *(unsigned*)&sK[key][d] = pack2(kv.x, kv.y);
    }
    // stage V transposed: one d from two consecutive keys -> one packed dword
    #pragma unroll
    for (int i = 0; i < 8; ++i) {
      int e  = tid + i * 128;           // 0..1023
      int kp = e >> 6, d = e & 63;      // key-pair 0..15, d 0..63
      const float* vp = &qkv[(rowbase + kb * 32 + 2 * kp) * 1536 + vcol + d];
      *(unsigned*)&sVt[d][2 * kp] = pack2(vp[0], vp[1536]);
    }
    __syncthreads();

    // S = Q K^T: load all 4 B-fragments first, then issue 4 WMMAs
    v16bf bk[2][2];
    #pragma unroll
    for (int sub = 0; sub < 2; ++sub)
      #pragma unroll
      for (int c = 0; c < 2; ++c) {
        const unsigned short* p = &sK[sub * 16 + n15][c * 32 + hg * 8];
        bk[sub][c] = ld_frag2(p, p + 16);
      }
    float s0[8], s1[8];
    #pragma unroll
    for (int sub = 0; sub < 2; ++sub) {
      v8f sf = {};
      sf = wmma_bf16(aq[0], bk[sub][0], sf);
      sf = wmma_bf16(aq[1], bk[sub][1], sf);
      #pragma unroll
      for (int r = 0; r < 8; ++r) {
        if (sub == 0) s0[r] = sf[r]; else s1[r] = sf[r];
      }
    }

    // causal mask + online softmax (row stats reduced within 16-lane groups)
    #pragma unroll
    for (int r = 0; r < 8; ++r) {
      int row = tile * 64 + wave * 16 + hg * 8 + r;
      int c0  = kb * 32 + n15;
      if (c0 > row)      s0[r] = -1e30f;
      if (c0 + 16 > row) s1[r] = -1e30f;
      float v = fmaxf(s0[r], s1[r]);
      #pragma unroll
      for (int off = 1; off < 16; off <<= 1) v = fmaxf(v, __shfl_xor(v, off, 32));
      float newm  = fmaxf(mstat[r], v);
      float alpha = __expf(mstat[r] - newm);
      float p0 = __expf(s0[r] - newm);
      float p1 = __expf(s1[r] - newm);
      float rsum = p0 + p1;
      #pragma unroll
      for (int off = 1; off < 16; off <<= 1) rsum += __shfl_xor(rsum, off, 32);
      lstat[r] = lstat[r] * alpha + rsum;
      mstat[r] = newm;
      #pragma unroll
      for (int nc = 0; nc < 4; ++nc) o[nc][r] *= alpha;
      sP[wave][hg * 8 + r][n15]      = f2bfu(p0);
      sP[wave][hg * 8 + r][16 + n15] = f2bfu(p1);
    }
    // same-wave LDS is processed in order (ISA 7.3); make it explicit and
    // block compiler reordering before reading P back as an A-fragment.
    asm volatile("s_wait_dscnt 0x0" ::: "memory");

    // O += P V: load P and all 4 V-fragments first, then issue 4 WMMAs
    v16bf ap = ld_frag2(&sP[wave][n15][hg * 8],
                        &sP[wave][n15][16 + hg * 8]);
    v16bf bv[4];
    #pragma unroll
    for (int nc = 0; nc < 4; ++nc) {
      const unsigned short* p = &sVt[nc * 16 + n15][hg * 8];
      bv[nc] = ld_frag2(p, p + 16);
    }
    #pragma unroll
    for (int nc = 0; nc < 4; ++nc)
      o[nc] = wmma_bf16(ap, bv[nc], o[nc]);
    __syncthreads();
  }

  // epilogue: O / l
  #pragma unroll
  for (int nc = 0; nc < 4; ++nc)
    #pragma unroll
    for (int r = 0; r < 8; ++r) {
      int t = tile * 64 + wave * 16 + hg * 8 + r;
      out[(rowbase + t) * LD_ + h * HD_ + nc * 16 + n15] = o[nc][r] / lstat[r];
    }
}

// ---------------------------------------------------------------------------
// Gate: logits = [local|global] @ w_gate(1024x2) + b; 2-way softmax; mix.
// One wave per token row; 8 rows per 256-thread block.
// ---------------------------------------------------------------------------
__global__ void __launch_bounds__(256)
gate_combine_kernel(const float* __restrict__ lout, const float* __restrict__ gout,
                    const float* __restrict__ w_gate, const float* __restrict__ b_gate,
                    float* __restrict__ out)
{
  const int tid  = threadIdx.x;
  const int wave = tid >> 5;
  const int lane = tid & 31;
  const int row  = blockIdx.x * 8 + wave;        // [0, B*T)
  const int b    = row >> 12;                    // T = 4096
  const int t    = row & (T_ - 1);
  const size_t lrow = (size_t)row * LD_;
  const size_t grow = ((size_t)b * TC_ + (t >> 2)) * LD_;  // repeat(R=4)

  float l0 = 0.f, l1 = 0.f;
  for (int c = lane; c < LD_; c += 32) {
    float lv = lout[lrow + c];
    float gv = gout[grow + c];
    l0 += lv * w_gate[2 * c]       + gv * w_gate[2 * (LD_ + c)];
    l1 += lv * w_gate[2 * c + 1]   + gv * w_gate[2 * (LD_ + c) + 1];
  }
  #pragma unroll
  for (int off = 1; off < 32; off <<= 1) {
    l0 += __shfl_xor(l0, off, 32);
    l1 += __shfl_xor(l1, off, 32);
  }
  l0 += b_gate[0]; l1 += b_gate[1];
  float m  = fmaxf(l0, l1);
  float e0 = __expf(l0 - m), e1 = __expf(l1 - m);
  float inv = 1.f / (e0 + e1);
  float g0 = e0 * inv, g1 = e1 * inv;

  const size_t orow = (size_t)row * E_;
  for (int c = lane; c < LD_; c += 32) {
    out[orow + c]       = g0 * lout[lrow + c];
    out[orow + LD_ + c] = g1 * gout[grow + c];
  }
}

// ---------------------------------------------------------------------------
extern "C" void kernel_launch(void* const* d_in, const int* in_sizes, int n_in,
                              void* d_out, int out_size, void* d_ws, size_t ws_size,
                              hipStream_t stream)
{
  (void)in_sizes; (void)n_in; (void)out_size; (void)ws_size;
  const float* x       = (const float*)d_in[0];
  const float* w_lqkv  = (const float*)d_in[1];
  const float* b_lqkv  = (const float*)d_in[2];
  const float* w_gqkv  = (const float*)d_in[3];
  const float* b_gqkv  = (const float*)d_in[4];
  const float* w_comp  = (const float*)d_in[5];
  const float* b_comp  = (const float*)d_in[6];
  const float* w_lproj = (const float*)d_in[7];
  const float* b_lproj = (const float*)d_in[8];
  const float* w_gproj = (const float*)d_in[9];
  const float* b_gproj = (const float*)d_in[10];
  const float* w_gate  = (const float*)d_in[11];
  const float* b_gate  = (const float*)d_in[12];
  float* out = (float*)d_out;

  // workspace layout (f32), ~104 MB total — L2-resident on MI455X (192 MB)
  float* ws = (float*)d_ws;
  const size_t BT  = (size_t)B_ * T_;    // 8192
  const size_t BTC = (size_t)B_ * TC_;   // 2048
  float* lqkv  = ws; ws += BT  * 1536;
  float* gqkv  = ws; ws += BTC * 1536;
  float* cg    = ws; ws += BTC * 512;
  float* lattn = ws; ws += BT  * 512;
  float* lprj  = ws; ws += BT  * 512;
  float* gattn = ws; ws += BTC * 512;
  float* gprj  = ws; ws += BTC * 512;

  dim3 blk(128);

  // local qkv: x[..., :512] (lda=1024) @ w_lqkv(512x1536)
  gemm_bias_kernel<<<dim3(1536 / 64, BT / 64), blk, 0, stream>>>(
      x, E_, w_lqkv, 1536, b_lqkv, lqkv, 1536, (int)BT, 1536, 512);

  // compressed (upper half only): x as (2048, 4096) @ w_comp[:, 512:] (ldw=1024)
  gemm_bias_kernel<<<dim3(512 / 64, BTC / 64), blk, 0, stream>>>(
      x, 4096, w_comp + 512, 1024, b_comp + 512, cg, 512, (int)BTC, 512, 4096);

  // global qkv: cg(2048x512) @ w_gqkv(512x1536)
  gemm_bias_kernel<<<dim3(1536 / 64, BTC / 64), blk, 0, stream>>>(
      cg, 512, w_gqkv, 1536, b_gqkv, gqkv, 1536, (int)BTC, 1536, 512);

  // causal attention, both resolutions
  flash_attn_kernel<<<dim3(B_ * HH_ * (T_ / 64)), blk, 0, stream>>>(lqkv, lattn, T_);
  flash_attn_kernel<<<dim3(B_ * HH_ * (TC_ / 64)), blk, 0, stream>>>(gqkv, gattn, TC_);

  // output projections
  gemm_bias_kernel<<<dim3(512 / 64, BT / 64), blk, 0, stream>>>(
      lattn, 512, w_lproj, 512, b_lproj, lprj, 512, (int)BT, 512, 512);
  gemm_bias_kernel<<<dim3(512 / 64, BTC / 64), blk, 0, stream>>>(
      gattn, 512, w_gproj, 512, b_gproj, gprj, 512, (int)BTC, 512, 512);

  // gate + combine
  gate_combine_kernel<<<dim3(BT / 8), dim3(256), 0, stream>>>(
      lprj, gprj, w_gate, b_gate, out);
}
